// MultiHeadAttention_85804856640142
// MI455X (gfx1250) — matrix operand order repaired and checked by reading, not compiled, not run on hardware
//
#include <hip/hip_runtime.h>

#define H   8
#define BB  8
#define SS  1024
#define DM  512
#define DK  64
#define BS  (BB * SS)

typedef __bf16 bf16_t;
typedef bf16_t        v16bf __attribute__((ext_vector_type(16)));
typedef bf16_t        v2bf  __attribute__((ext_vector_type(2)));
typedef float         v8f   __attribute__((ext_vector_type(8)));
typedef unsigned int  v8u   __attribute__((ext_vector_type(8)));

// native pack of two f32 -> packed bf16 pair (backend can use v_cvt_pk_bf16_f32)
static __device__ __forceinline__ unsigned int pack2(float a, float b) {
  v2bf t;
  t.x = (bf16_t)a;
  t.y = (bf16_t)b;
  return __builtin_bit_cast(unsigned int, t);
}
static __device__ __forceinline__ v8u join8(uint4 lo, uint4 hi) {
  v8u r = {lo.x, lo.y, lo.z, lo.w, hi.x, hi.y, hi.z, hi.w};
  return r;
}
static __device__ __forceinline__ v16bf cast16(v8u u) {
  return __builtin_bit_cast(v16bf, u);
}
static __device__ __forceinline__ v8f wmma_bf16(v16bf a, v16bf b, v8f c) {
  // (neg_a, A, neg_b, B, c_mod, C, reuse_a, reuse_b)
  return __builtin_amdgcn_wmma_f32_16x16x32_bf16(false, a, false, b, (short)0, c,
                                                 false, false);
}

// ---------------------------------------------------------------------------
// Kernel 0: one-time operand conditioning.
//  - w_qs/w_ks/w_vs f32 [h][k][n] -> wt bf16 [which][h][n][k]  (transposed)
//  - proj_w f32 [m][c]            -> pwb bf16 [m][c]
// ---------------------------------------------------------------------------
__global__ __launch_bounds__(256)
void k_prep(const float* __restrict__ wq, const float* __restrict__ wk,
            const float* __restrict__ wv, const float* __restrict__ pw,
            bf16_t* __restrict__ wt, bf16_t* __restrict__ pwb) {
  const int idx = blockIdx.x * 256 + threadIdx.x;
  const int NW = 3 * H * DK * DM;
  if (idx < NW) {
    const int kk   = idx % DM;
    const int n    = (idx / DM) % DK;
    const int h    = (idx / (DM * DK)) % H;
    const int wsel = idx / (DM * DK * H);
    const float* src = (wsel == 0) ? wq : (wsel == 1) ? wk : wv;
    wt[idx] = (bf16_t)src[((size_t)h * DM + kk) * DK + n];
  } else {
    const int o = idx - NW;                 // < DM*DM by grid construction
    pwb[o] = (bf16_t)pw[o];
  }
}

// ---------------------------------------------------------------------------
// Kernel 1: per-head Q/K/V projections (16x512)x(512x64) per wave.
// A from f32 activations (b128 loads + pack), B from bf16 wt (pure b128).
// Q,K stored row-major bf16; V stored transposed [dv][kpos] bf16 for aV.
// ---------------------------------------------------------------------------
__global__ __launch_bounds__(32)
void k_qkv_proj(const float* __restrict__ q, const float* __restrict__ k,
                const float* __restrict__ v, const bf16_t* __restrict__ wt,
                bf16_t* __restrict__ qs, bf16_t* __restrict__ ks,
                bf16_t* __restrict__ vt) {
  const int lane = threadIdx.x & 31;
  const int mrow = lane & 15;
  const int kg   = lane >> 4;               // K-half select (ISA A/B layouts)
  int id = blockIdx.x;
  const int which = id / (H * (BS / 16));
  id %= H * (BS / 16);
  const int h  = id / (BS / 16);
  const int rt = id % (BS / 16);

  const float* x    = (which == 0) ? q : (which == 1) ? k : v;
  const float* xrow = x + (size_t)(rt * 16 + mrow) * DM;
  const bf16_t* wbase = wt + (size_t)(which * H + h) * DK * DM;

  v8f acc0 = {}, acc1 = {}, acc2 = {}, acc3 = {};
  for (int kc = 0; kc < DM; kc += 32) {
    const float4 r0 = *(const float4*)(xrow + kc + kg * 8);
    const float4 r1 = *(const float4*)(xrow + kc + kg * 8 + 4);
    const float4 r2 = *(const float4*)(xrow + kc + 16 + kg * 8);
    const float4 r3 = *(const float4*)(xrow + kc + 16 + kg * 8 + 4);
    v8u au = { pack2(r0.x, r0.y), pack2(r0.z, r0.w),
               pack2(r1.x, r1.y), pack2(r1.z, r1.w),
               pack2(r2.x, r2.y), pack2(r2.z, r2.w),
               pack2(r3.x, r3.y), pack2(r3.z, r3.w) };
    const v16bf a = cast16(au);
#pragma unroll
    for (int nt = 0; nt < 4; ++nt) {
      const bf16_t* wn = wbase + (size_t)(nt * 16 + mrow) * DM + kc + kg * 8;
      const v16bf bf = cast16(join8(*(const uint4*)wn, *(const uint4*)(wn + 16)));
      if      (nt == 0) acc0 = wmma_bf16(a, bf, acc0);
      else if (nt == 1) acc1 = wmma_bf16(a, bf, acc1);
      else if (nt == 2) acc2 = wmma_bf16(a, bf, acc2);
      else              acc3 = wmma_bf16(a, bf, acc3);
    }
  }

  if (which < 2) {
    bf16_t* outp = ((which == 0) ? qs : ks) + (size_t)h * BS * DK
                   + (size_t)rt * 16 * DK;
#pragma unroll
    for (int r = 0; r < 8; ++r) {
      const int m = r + kg * 8;
      outp[m * DK +  0 + mrow] = (bf16_t)acc0[r];
      outp[m * DK + 16 + mrow] = (bf16_t)acc1[r];
      outp[m * DK + 32 + mrow] = (bf16_t)acc2[r];
      outp[m * DK + 48 + mrow] = (bf16_t)acc3[r];
    }
  } else {
    const int bidx = (rt * 16) >> 10;       // 16-row tile never crosses batch
    const int q0   = (rt * 16) & (SS - 1);
    bf16_t* vb = vt + (size_t)(h * BB + bidx) * DK * SS;
#pragma unroll
    for (int r = 0; r < 8; ++r) {
      const int qpos = q0 + r + kg * 8;
      vb[(size_t)( 0 + mrow) * SS + qpos] = (bf16_t)acc0[r];
      vb[(size_t)(16 + mrow) * SS + qpos] = (bf16_t)acc1[r];
      vb[(size_t)(32 + mrow) * SS + qpos] = (bf16_t)acc2[r];
      vb[(size_t)(48 + mrow) * SS + qpos] = (bf16_t)acc3[r];
    }
  }
}

// ---------------------------------------------------------------------------
// Kernel 2: one wave per (h, b, 16-row q-tile).
// scores (WMMA) -> mask/scale -> LDS -> softmax (f32, float4) -> attns,
// then alpha @ V^T via WMMA into bf16 concat buffer.
// ---------------------------------------------------------------------------
__global__ __launch_bounds__(32)
void k_attention(const bf16_t* __restrict__ qs, const bf16_t* __restrict__ ks,
                 const bf16_t* __restrict__ vt, const int* __restrict__ adj,
                 float* __restrict__ attns, bf16_t* __restrict__ catb) {
  __shared__ float sc[16][SS];              // 64 KB score rows
  const int lane = threadIdx.x & 31;
  const int mrow = lane & 15;
  const int kg   = lane >> 4;
  int id = blockIdx.x;
  const int h = id / (BB * (SS / 16));
  id %= BB * (SS / 16);
  const int b  = id / (SS / 16);
  const int qt = id % (SS / 16);

  const bf16_t* qrow  = qs + ((size_t)(h * BB + b) * SS + qt * 16 + mrow) * DK;
  const bf16_t* kbase = ks + (size_t)(h * BB + b) * SS * DK;
  const bf16_t* vbase = vt + (size_t)(h * BB + b) * DK * SS;

  // Q fragments pinned in VGPRs (4 b128 loads total)
  const v16bf qa0 = cast16(join8(*(const uint4*)(qrow + kg * 8),
                                 *(const uint4*)(qrow + 16 + kg * 8)));
  const v16bf qa1 = cast16(join8(*(const uint4*)(qrow + 32 + kg * 8),
                                 *(const uint4*)(qrow + 48 + kg * 8)));

  for (int kt = 0; kt < SS / 16; ++kt) {
    const bf16_t* krow = kbase + (size_t)(kt * 16 + mrow) * DK;
    if (kt + 1 < SS / 16)
      __builtin_prefetch(kbase + (size_t)((kt + 1) * 16 + mrow) * DK, 0, 3);
    const v16bf b0 = cast16(join8(*(const uint4*)(krow + kg * 8),
                                  *(const uint4*)(krow + 16 + kg * 8)));
    const v16bf b1 = cast16(join8(*(const uint4*)(krow + 32 + kg * 8),
                                  *(const uint4*)(krow + 48 + kg * 8)));
    v8f acc = {};
    acc = wmma_bf16(qa0, b0, acc);
    acc = wmma_bf16(qa1, b1, acc);

    const int col = kt * 16 + mrow;
#pragma unroll
    for (int r = 0; r < 8; ++r) {
      const int m = r + kg * 8;
      const int a = adj[((size_t)b * SS + qt * 16 + m) * SS + col];
      sc[m][col] = (a > 0) ? acc[r] * 32.0f : -1.0e30f;  // scale = sqrt(1024)
    }
  }
  __syncthreads();

  // softmax: 2 lanes per row, float4 throughout, wave32 xor reduction
  {
    const int row  = lane >> 1;
    const int base = (lane & 1) * (SS / 2);
    float4* s4 = (float4*)&sc[row][base];
    float mx = -3.0e38f;
    for (int c = 0; c < SS / 8; ++c) {
      const float4 t = s4[c];
      mx = fmaxf(mx, fmaxf(fmaxf(t.x, t.y), fmaxf(t.z, t.w)));
    }
    mx = fmaxf(mx, __shfl_xor(mx, 1, 32));
    float sum = 0.0f;
    for (int c = 0; c < SS / 8; ++c) {
      float4 t = s4[c];
      t.x = __expf(t.x - mx); t.y = __expf(t.y - mx);
      t.z = __expf(t.z - mx); t.w = __expf(t.w - mx);
      s4[c] = t;
      sum += (t.x + t.y) + (t.z + t.w);
    }
    sum += __shfl_xor(sum, 1, 32);
    const float inv = 1.0f / sum;
    float4* arow = (float4*)(attns + ((size_t)(h * BB + b) * SS + qt * 16 + row) * SS
                             + base);
    for (int c = 0; c < SS / 8; ++c) {
      float4 t = s4[c];
      t.x *= inv; t.y *= inv; t.z *= inv; t.w *= inv;
      s4[c]   = t;    // alpha kept in LDS for aV
      arow[c] = t;    // stream alpha straight to d_out (attns)
    }
  }
  __syncthreads();

  // out_head = alpha (16 x 1024) @ V (1024 x 64), V^T gives contiguous B frags
  v8f o0 = {}, o1 = {}, o2 = {}, o3 = {};
  for (int kc = 0; kc < SS; kc += 32) {
    const float4 a0 = *(const float4*)&sc[mrow][kc + kg * 8];
    const float4 a1 = *(const float4*)&sc[mrow][kc + kg * 8 + 4];
    const float4 a2 = *(const float4*)&sc[mrow][kc + 16 + kg * 8];
    const float4 a3 = *(const float4*)&sc[mrow][kc + 16 + kg * 8 + 4];
    v8u au = { pack2(a0.x, a0.y), pack2(a0.z, a0.w),
               pack2(a1.x, a1.y), pack2(a1.z, a1.w),
               pack2(a2.x, a2.y), pack2(a2.z, a2.w),
               pack2(a3.x, a3.y), pack2(a3.z, a3.w) };
    const v16bf a = cast16(au);
#pragma unroll
    for (int nt = 0; nt < 4; ++nt) {
      const bf16_t* vr = vbase + (size_t)(nt * 16 + mrow) * SS + kc + kg * 8;
      const v16bf bf = cast16(join8(*(const uint4*)vr, *(const uint4*)(vr + 16)));
      if      (nt == 0) o0 = wmma_bf16(a, bf, o0);
      else if (nt == 1) o1 = wmma_bf16(a, bf, o1);
      else if (nt == 2) o2 = wmma_bf16(a, bf, o2);
      else              o3 = wmma_bf16(a, bf, o3);
    }
  }
  // concat layout [b][q][h*64+dv], stored bf16 for the final WMMA GEMM
  bf16_t* crow = catb + ((size_t)b * SS + qt * 16) * (H * DK) + h * DK;
#pragma unroll
  for (int r = 0; r < 8; ++r) {
    const int m = r + kg * 8;
    crow[(size_t)m * (H * DK) +  0 + mrow] = (bf16_t)o0[r];
    crow[(size_t)m * (H * DK) + 16 + mrow] = (bf16_t)o1[r];
    crow[(size_t)m * (H * DK) + 32 + mrow] = (bf16_t)o2[r];
    crow[(size_t)m * (H * DK) + 48 + mrow] = (bf16_t)o3[r];
  }
}

// ---------------------------------------------------------------------------
// Kernel 3: out = concat @ proj_w^T + b.  All-bf16 operands: zero conversion
// VALU in the loop — just b128 loads + 16 WMMAs per wave.
// ---------------------------------------------------------------------------
__global__ __launch_bounds__(32)
void k_proj(const bf16_t* __restrict__ catb, const bf16_t* __restrict__ pwb,
            const float* __restrict__ pb, float* __restrict__ out) {
  const int lane = threadIdx.x & 31;
  const int mrow = lane & 15;
  const int kg   = lane >> 4;
  const int it = blockIdx.x / (DM / 16);
  const int mt = blockIdx.x % (DM / 16);

  const bf16_t* arow = catb + (size_t)(it * 16 + mrow) * DM;
  const bf16_t* wrow = pwb  + (size_t)(mt * 16 + mrow) * DM;  // B[k][n]=pw[n][k]

  v8f acc = {};
  for (int kc = 0; kc < DM; kc += 32) {
    const v16bf a  = cast16(join8(*(const uint4*)(arow + kc + kg * 8),
                                  *(const uint4*)(arow + kc + 16 + kg * 8)));
    const v16bf bf = cast16(join8(*(const uint4*)(wrow + kc + kg * 8),
                                  *(const uint4*)(wrow + kc + 16 + kg * 8)));
    acc = wmma_bf16(a, bf, acc);
  }
  const int n = mt * 16 + mrow;
  const float bias = pb[n];
#pragma unroll
  for (int r = 0; r < 8; ++r)
    out[(size_t)(it * 16 + r + kg * 8) * DM + n] = acc[r] + bias;
}

// ---------------------------------------------------------------------------
extern "C" void kernel_launch(void* const* d_in, const int* in_sizes, int n_in,
                              void* d_out, int out_size, void* d_ws, size_t ws_size,
                              hipStream_t stream) {
  (void)in_sizes; (void)n_in; (void)out_size; (void)ws_size;
  const float* q   = (const float*)d_in[0];
  const float* k   = (const float*)d_in[1];
  const float* v   = (const float*)d_in[2];
  const int*   adj = (const int*)  d_in[3];
  const float* wq  = (const float*)d_in[4];
  const float* wk  = (const float*)d_in[5];
  const float* wv  = (const float*)d_in[6];
  const float* pw  = (const float*)d_in[7];
  const float* pb  = (const float*)d_in[8];

  float* out   = (float*)d_out;
  float* attns = out + (size_t)BB * SS * DM;          // tuple order: (out, attns)

  bf16_t* ws   = (bf16_t*)d_ws;
  bf16_t* qs   = ws;                                   // 8 MB
  bf16_t* ksb  = qs  + (size_t)H * BS * DK;            // 8 MB
  bf16_t* vtb  = ksb + (size_t)H * BS * DK;            // 8 MB (V transposed)
  bf16_t* catb = vtb + (size_t)H * BS * DK;            // 8 MB (concat, bf16)
  bf16_t* wtb  = catb + (size_t)BS * DM;               // 1.5 MB (weights^T)
  bf16_t* pwb  = wtb + (size_t)3 * H * DK * DM;        // 0.5 MB (proj_w bf16)

  k_prep<<<(3 * H * DK * DM + DM * DM) / 256, 256, 0, stream>>>(wq, wk, wv, pw,
                                                                wtb, pwb);
  k_qkv_proj<<<3 * H * (BS / 16), 32, 0, stream>>>(q, k, v, wtb, qs, ksb, vtb);
  k_attention<<<H * BB * (SS / 16), 32, 0, stream>>>(qs, ksb, vtb, adj, attns, catb);
  k_proj<<<(BS / 16) * (DM / 16), 32, 0, stream>>>(catb, pwb, pb, out);
}
